// LstmAutoEncoder_50775103373426
// MI455X (gfx1250) — matrix operand order repaired and checked by reading, not compiled, not run on hardware
//
#include <hip/hip_runtime.h>
#include <hip/hip_bf16.h>
#include <stdint.h>

// ---------------------------------------------------------------------------
// LSTM autoencoder on gfx1250 (MI455X): persistent per-batch-tile kernel,
// all GEMMs via V_WMMA_F32_16X16X32_BF16, weights pre-packed into per-wave
// B-fragment order, h/x double-buffered in LDS, c held in registers.
// ---------------------------------------------------------------------------

#define POSE_DIM 96
#define H_DIM    512
#define SEQ_LEN  50
#define DEC_LEN  25
#define BATCH    2048

#define KROW     (H_DIM + POSE_DIM)   // 608 = combined [h | x] reduction dim
#define KB_GATE  (KROW / 32)          // 19 K-blocks of 32
#define NB_GATE  (4 * H_DIM / 16)     // 128 N-tiles of 16 (i,f,g,o each 32)
#define KB_PROJ  (H_DIM / 32)         // 16
#define NB_PROJ  (POSE_DIM / 16)      // 6

#define MT       16                   // batch rows per block (one M-tile)
#define NWAVES   8
#define NTHREADS (NWAVES * 32)
#define NBLOCKS  (BATCH / MT)         // 128
#define TPW      (32 / NWAVES)        // hidden-block tasks per wave = 4

typedef __attribute__((ext_vector_type(16))) __bf16 v16bf;
typedef __attribute__((ext_vector_type(8)))  __bf16 v8bf;
typedef __attribute__((ext_vector_type(8)))  float  v8f;

#define ENC_FRAG_ELEMS ((size_t)KB_GATE * NB_GATE * 512)   // 1,245,184 bf16
#define PROJ_FRAG_ELEMS ((size_t)KB_PROJ * NB_PROJ * 512)  //    49,152 bf16
// d_ws layout (bytes): [enc_Wc | dec_Wc | proj_Wc]  ~4.85 MB total

__device__ __forceinline__ float sigm(float x) {
  return 1.0f / (1.0f + __expf(-x));
}

__device__ __forceinline__ float tanh_fast(float x) {
  // numerically safe fast tanh via __expf; avoids inf/inf for large |x|
  float t = __expf(-2.0f * __builtin_fabsf(x));
  float r = (1.0f - t) / (1.0f + t);
  return __builtin_copysignf(r, x);
}

// ---------------------------------------------------------------------------
// Pack kernel: build bf16 combined weight Wc[k][n] = (k<kmain? Wmain[n][k]
// : Wtail[n][k-kmain]) into per-wave WMMA B-fragments. Fragment (kb, nb)
// holds a 32x16 (KxN) tile; lane l stores 16 contiguous bf16:
//   elem t of lane l = Wc[kb*32 + (l>>4)*8 + (t&7) + (t>>3)*16][nb*16 + (l&15)]
// so the compute kernel loads one v16bf (32B) per lane, fully coalesced.
// ---------------------------------------------------------------------------
__global__ void pack_frags(const float* __restrict__ Wmain, int kmain,
                           const float* __restrict__ Wtail, int ktail,
                           int NB, int KB, __bf16* __restrict__ dst) {
  int total = KB * NB * 512;
  for (int e = blockIdx.x * blockDim.x + threadIdx.x; e < total;
       e += gridDim.x * blockDim.x) {
    int f  = e >> 9;       // fragment index
    int r  = e & 511;
    int l  = r >> 4;       // lane 0..31
    int tt = r & 15;       // element within lane
    int kb = f / NB, nb = f % NB;
    int k  = kb * 32 + ((l >> 4) << 3) + (tt & 7) + ((tt >> 3) << 4);
    int n  = nb * 16 + (l & 15);
    float v = (k < kmain) ? Wmain[(size_t)n * kmain + k]
                          : Wtail[(size_t)n * ktail + (k - kmain)];
    dst[e] = (__bf16)v;
  }
}

// ---------------------------------------------------------------------------
// Persistent LSTM kernel: one block = 16 batch rows, 8 waves, 75 timesteps.
// ---------------------------------------------------------------------------
__global__ __launch_bounds__(NTHREADS)
void lstm_ae_kernel(const float* __restrict__ x,
                    const v16bf* __restrict__ wenc,   // packed enc fragments
                    const float* __restrict__ enc_b,
                    const v16bf* __restrict__ wdec,   // packed dec fragments
                    const float* __restrict__ dec_b,
                    const v16bf* __restrict__ wproj,  // packed proj fragments
                    const float* __restrict__ proj_b,
                    float* __restrict__ out) {
  // double-buffered combined input [h(512) | x(96)] in bf16, row stride 608
  __shared__ __align__(16) __bf16 s_in[2][MT][KROW];   // 38,912 B

  const int tid    = threadIdx.x;
  const int lane   = tid & 31;
  const int wave   = tid >> 5;
  const int batch0 = blockIdx.x * MT;

  // c state in registers: TPW tasks per wave, 8 accumulator rows per lane
  float cst[TPW][8] = {};

  // ---- init: zero h portion of buf0, load frame 0 into x portion ----
  for (int idx = tid; idx < MT * H_DIM; idx += NTHREADS)
    s_in[0][idx / H_DIM][idx % H_DIM] = (__bf16)0.0f;
  for (int idx = tid; idx < MT * POSE_DIM; idx += NTHREADS) {
    int m = idx / POSE_DIM, n = idx % POSE_DIM;
    s_in[0][m][H_DIM + n] =
        (__bf16)x[((size_t)(batch0 + m) * SEQ_LEN + 0) * POSE_DIM + n];
  }
  __syncthreads();

  // A-fragment loader: 16x32 bf16 tile of [h|x] per the CDNA5 A layout.
  // lane l: row m = l&15; half = l>>4; K = kb*32 + half*8 + {0..7, 16..23}
  auto load_a = [&](int buf, int kb) -> v16bf {
    const __bf16* rp = &s_in[buf][lane & 15][kb * 32 + ((lane >> 4) << 3)];
    v8bf lo = *(const v8bf*)rp;
    v8bf hi = *(const v8bf*)(rp + 16);
    return __builtin_shufflevector(lo, hi, 0, 1, 2, 3, 4, 5, 6, 7,
                                           8, 9, 10, 11, 12, 13, 14, 15);
  };

  int cur = 0;
  for (int t = 0; t < SEQ_LEN + DEC_LEN; ++t) {
    const bool   enc  = (t < SEQ_LEN);
    const v16bf* wv   = enc ? wenc : wdec;
    const float* bias = enc ? enc_b : dec_b;
    const int    nxt  = cur ^ 1;

    // ---- gate phase: gates(16x2048) = [h|x] @ Wc, 4 gate tiles per task ----
    #pragma unroll
    for (int slot = 0; slot < TPW; ++slot) {
      const int nh = slot * NWAVES + wave;   // hidden 16-block 0..31
      v8f acc[4] = {};
      for (int kb = 0; kb < KB_GATE; ++kb) {
        v16bf a = load_a(cur, kb);
        #pragma unroll
        for (int g = 0; g < 4; ++g) {
          v16bf b = wv[((size_t)kb * NB_GATE + g * 32 + nh) * 32 + lane];
          acc[g] = __builtin_amdgcn_wmma_f32_16x16x32_bf16(
              false, a, false, b, (short)0, acc[g], false, false);
        }
      }
      // per-lane cell update; C layout: lane<16 -> m=j, lane>=16 -> m=8+j,
      // n = lane&15  (identical slot across the four gate accumulators)
      const int nloc = lane & 15;
      const int col  = nh * 16 + nloc;
      const float bi = bias[0 * H_DIM + col];
      const float bf = bias[1 * H_DIM + col];
      const float bg = bias[2 * H_DIM + col];
      const float bo = bias[3 * H_DIM + col];
      #pragma unroll
      for (int j = 0; j < 8; ++j) {
        const int m = ((lane >> 4) << 3) + j;
        float iv = acc[0][j] + bi;
        float fv = acc[1][j] + bf;
        float gv = acc[2][j] + bg;
        float ov = acc[3][j] + bo;
        float cn = sigm(fv) * cst[slot][j] + sigm(iv) * tanh_fast(gv);
        cst[slot][j] = cn;
        float hn = sigm(ov) * tanh_fast(cn);
        s_in[nxt][m][col] = (__bf16)hn;
      }
    }

    if (enc) {
      // stage next frame (last frame repeats as first decoder input)
      const int f = (t + 1 < SEQ_LEN) ? t + 1 : SEQ_LEN - 1;
      for (int idx = tid; idx < MT * POSE_DIM; idx += NTHREADS) {
        int m = idx / POSE_DIM, n = idx % POSE_DIM;
        s_in[nxt][m][H_DIM + n] =
            (__bf16)x[((size_t)(batch0 + m) * SEQ_LEN + f) * POSE_DIM + n];
      }
      __syncthreads();
    } else {
      __syncthreads();   // h_new complete in s_in[nxt]
      // ---- projection: y(16x96) = h_new @ proj_W^T + proj_b ----
      if (wave < NB_PROJ) {
        const int nt = wave;
        v8f accp = {};
        for (int kb = 0; kb < KB_PROJ; ++kb) {
          v16bf a = load_a(nxt, kb);   // reads h portion only (kb < 16)
          v16bf b = wproj[((size_t)kb * NB_PROJ + nt) * 32 + lane];
          accp = __builtin_amdgcn_wmma_f32_16x16x32_bf16(
              false, a, false, b, (short)0, accp, false, false);
        }
        const int n  = nt * 16 + (lane & 15);
        const float pb = proj_b[n];
        const int d  = t - SEQ_LEN;
        #pragma unroll
        for (int j = 0; j < 8; ++j) {
          const int m = ((lane >> 4) << 3) + j;
          float yv = accp[j] + pb;
          out[((size_t)(batch0 + m) * DEC_LEN + d) * POSE_DIM + n] = yv;
          s_in[nxt][m][H_DIM + n] = (__bf16)yv;   // feedback input
        }
      }
      __syncthreads();   // y visible before next step reads it
    }
    cur = nxt;
  }
}

// ---------------------------------------------------------------------------
// Host launcher. Input order: x, enc_Wih, enc_Whh, enc_b, dec_Wih, dec_Whh,
// dec_b, proj_W, proj_b.  Needs ~5.1 MB of d_ws for packed bf16 weights.
// ---------------------------------------------------------------------------
extern "C" void kernel_launch(void* const* d_in, const int* in_sizes, int n_in,
                              void* d_out, int out_size, void* d_ws, size_t ws_size,
                              hipStream_t stream) {
  (void)in_sizes; (void)n_in; (void)out_size; (void)ws_size;
  const float* x       = (const float*)d_in[0];
  const float* enc_Wih = (const float*)d_in[1];
  const float* enc_Whh = (const float*)d_in[2];
  const float* enc_b   = (const float*)d_in[3];
  const float* dec_Wih = (const float*)d_in[4];
  const float* dec_Whh = (const float*)d_in[5];
  const float* dec_b   = (const float*)d_in[6];
  const float* proj_W  = (const float*)d_in[7];
  const float* proj_b  = (const float*)d_in[8];

  char* ws = (char*)d_ws;
  __bf16* wenc  = (__bf16*)(ws);
  __bf16* wdec  = (__bf16*)(ws + ENC_FRAG_ELEMS * sizeof(__bf16));
  __bf16* wproj = (__bf16*)(ws + 2 * ENC_FRAG_ELEMS * sizeof(__bf16));

  // pack combined [Whh^T ; Wih^T] gate weights + proj weights into fragments
  pack_frags<<<1024, 256, 0, stream>>>(enc_Whh, H_DIM, enc_Wih, POSE_DIM,
                                       NB_GATE, KB_GATE, wenc);
  pack_frags<<<1024, 256, 0, stream>>>(dec_Whh, H_DIM, dec_Wih, POSE_DIM,
                                       NB_GATE, KB_GATE, wdec);
  pack_frags<<<256, 256, 0, stream>>>(proj_W, H_DIM, nullptr, 1,
                                      NB_PROJ, KB_PROJ, wproj);

  lstm_ae_kernel<<<NBLOCKS, NTHREADS, 0, stream>>>(
      x, (const v16bf*)wenc, enc_b, (const v16bf*)wdec, dec_b,
      (const v16bf*)wproj, proj_b, (float*)d_out);
}
